// CNNtoGraph_77936476553433
// MI455X (gfx1250) — matrix operand order repaired and checked by reading, not compiled, not run on hardware
//
#include <hip/hip_runtime.h>
#include <stdint.h>

// ---------------------------------------------------------------------------
// Problem constants (from the reference)
// ---------------------------------------------------------------------------
#define B_  2048
#define N_  6
#define D_  2048
#define H_  1024
#define C_  200
#define K1  (2 * D_)      // 4096  : K of GEMM1
#define CP  256           // padded classifier N
#define ALPHA_ 0.015f
#define SCALE_ 24.0f

// ---------------------------------------------------------------------------
// CDNA5 WMMA types
// ---------------------------------------------------------------------------
typedef __attribute__((ext_vector_type(16))) __bf16 v16bf;
typedef __attribute__((ext_vector_type(8)))  float  v8f;

union Frag {
    uint4 u[2];
    v16bf v;
};

// float -> bf16 (round-to-nearest-even), pure integer ops
__device__ __forceinline__ uint16_t f2bf(float f) {
    uint32_t u = __builtin_bit_cast(uint32_t, f);
    u += 0x7fffu + ((u >> 16) & 1u);
    return (uint16_t)(u >> 16);
}

// ---------------------------------------------------------------------------
// Kernel 1: per-batch row-sum coefficients of the spatial weight matrix.
// coef[b][u] = (SCALE/30) * sum_{v != u} max(w0[u][v] - mean_w, 0)
// ---------------------------------------------------------------------------
__global__ void coef_kernel(const float* __restrict__ cdds, float* __restrict__ coef) {
    int b = blockIdx.x * blockDim.x + threadIdx.x;
    if (b >= B_) return;
    const float* c = cdds + (size_t)b * N_ * 6;

    float cx[N_], cy[N_];
#pragma unroll
    for (int u = 0; u < N_; ++u) {
        cy[u] = (c[u * 6 + 1] + c[u * 6 + 3]) * 0.5f;
        cx[u] = (c[u * 6 + 2] + c[u * 6 + 4]) * 0.5f;
    }
    float w0[N_][N_];
    float s = 0.f;
#pragma unroll
    for (int u = 0; u < N_; ++u) {
#pragma unroll
        for (int v = 0; v < N_; ++v) {
            if (u == v) { w0[u][v] = 0.f; continue; }
            float dx = cx[u] - cx[v];
            float dy = cy[u] - cy[v];
            float d  = __builtin_sqrtf(dx * dx + dy * dy);
            float w  = __expf(-ALPHA_ * d);
            w0[u][v] = w;
            s += w;
        }
    }
    float mean_w = s * (1.0f / (N_ * (N_ - 1)));
#pragma unroll
    for (int u = 0; u < N_; ++u) {
        float rs = 0.f;
#pragma unroll
        for (int v = 0; v < N_; ++v) {
            if (v == u) continue;
            rs += fmaxf(w0[u][v] - mean_w, 0.f);
        }
        // SCALE * rowsum, then /5 (hidden) and /6 (mean over nodes)
        coef[b * N_ + u] = rs * (SCALE_ / ((N_ - 1) * (float)N_));
    }
}

// ---------------------------------------------------------------------------
// Kernel 2: stream part_feats once; build Q (B x 4096, bf16):
//   Q[b][k]      = sum_u coef[b][u] * pf[b][u][k]        (weighted "hidden" part)
//   Q[b][D+k]    = (1/6) * sum_u pf[b][u][k]             (mean part)
// ---------------------------------------------------------------------------
__global__ void build_q(const float* __restrict__ pf, const float* __restrict__ coef,
                        uint16_t* __restrict__ Q) {
    int idx = blockIdx.x * blockDim.x + threadIdx.x;   // over B*D
    int b = idx >> 11;           // /D_
    int k = idx & (D_ - 1);
    const float* p = pf + ((size_t)b * N_) * D_ + k;
    const float* cf = coef + b * N_;
    float q1 = 0.f, q2 = 0.f;
#pragma unroll
    for (int u = 0; u < N_; ++u) {
        float x = p[(size_t)u * D_];
        q1 = fmaf(cf[u], x, q1);
        q2 += x;
    }
    q2 *= (1.0f / N_);
    uint16_t* row = Q + (size_t)b * K1;
    row[k]      = f2bf(q1);
    row[D_ + k] = f2bf(q2);
}

// ---------------------------------------------------------------------------
// Kernel 3: LDS-tiled transpose + bf16 convert of fc_w (4096 x 1024)
//           -> fcwT (1024 x 4096) so GEMM1 B-fragments are contiguous.
// ---------------------------------------------------------------------------
__global__ void transpose_fcw(const float* __restrict__ W, uint16_t* __restrict__ WT) {
    __shared__ float tile[32][33];
    int k0 = blockIdx.x * 32;
    int n0 = blockIdx.y * 32;
    int tx = threadIdx.x;        // 0..31
    int ty = threadIdx.y;        // 0..7
#pragma unroll
    for (int i = 0; i < 4; ++i)
        tile[ty + i * 8][tx] = W[(size_t)(k0 + ty + i * 8) * H_ + (n0 + tx)];
    __syncthreads();
#pragma unroll
    for (int i = 0; i < 4; ++i)
        WT[(size_t)(n0 + ty + i * 8) * K1 + (k0 + tx)] = f2bf(tile[tx][ty + i * 8]);
}

// ---------------------------------------------------------------------------
// Kernel 4: transpose + zero-pad cls_w (1024 x 200) -> clsT (256 x 1024, bf16)
// ---------------------------------------------------------------------------
__global__ void build_clsT(const float* __restrict__ W, uint16_t* __restrict__ WT) {
    int idx = blockIdx.x * blockDim.x + threadIdx.x;   // over CP*H_
    int n = idx >> 10;           // /H_
    int k = idx & (H_ - 1);
    float v = (n < C_) ? W[(size_t)k * C_ + n] : 0.f;
    WT[(size_t)n * H_ + k] = f2bf(v);
}

// ---------------------------------------------------------------------------
// Kernel 5: GEMM1  g_ft = Q (2048 x 4096, bf16) @ fc_w + fc_b -> gft bf16
// WG: 8 waves (4 along M, 2 along N); wave tile 16 x 64 (4 accumulators).
// v_wmma_f32_16x16x32_bf16 per the CDNA5 A/B VGPR layouts.
// ---------------------------------------------------------------------------
__global__ __launch_bounds__(256) void gemm1(const uint16_t* __restrict__ A,   // Q
                                             const uint16_t* __restrict__ BT,  // fcwT
                                             const float* __restrict__ bias,   // fc_b
                                             uint16_t* __restrict__ Cout) {    // gft bf16
    const int lane = threadIdx.x & 31;
    const int wave = threadIdx.x >> 5;
    const int wm = wave & 3;                 // 0..3
    const int wn = wave >> 2;                // 0..1
    const int m0 = blockIdx.x * 64 + wm * 16;
    const int n0 = blockIdx.y * 128 + wn * 64;
    const int l15 = lane & 15;
    const bool hi = lane >= 16;

    v8f acc[4] = {};
    const uint16_t* arow = A + (size_t)(m0 + l15) * K1;

    for (int k0 = 0; k0 < K1; k0 += 32) {
        Frag a;
        // A 16x32 bf16 layout: lanes 0-15 -> K{0..7,16..23}, lanes 16-31 -> K{8..15,24..31}
        int ka = k0 + (hi ? 8 : 0);
        a.u[0] = *(const uint4*)(arow + ka);
        a.u[1] = *(const uint4*)(arow + ka + 16);
#pragma unroll
        for (int t = 0; t < 4; ++t) {
            // B 32x16 bf16 layout: lane l<16 -> (col=l, K{0..7,16..23}), lane l>=16 -> (col, K{8..15,24..31})
            const uint16_t* bcol = BT + (size_t)(n0 + t * 16 + l15) * K1 + k0 + (hi ? 8 : 0);
            Frag bf;
            bf.u[0] = *(const uint4*)(bcol);
            bf.u[1] = *(const uint4*)(bcol + 16);
            acc[t] = __builtin_amdgcn_wmma_f32_16x16x32_bf16(
                false, a.v, false, bf.v, (short)0, acc[t], false, false);
        }
    }

    // Epilogue: + fc_b, convert to bf16, store g_ft
    const int rbase = m0 + (hi ? 8 : 0);
#pragma unroll
    for (int t = 0; t < 4; ++t) {
        int col = n0 + t * 16 + l15;
        float bv = bias[col];
#pragma unroll
        for (int r = 0; r < 8; ++r) {
            float val = acc[t][r] + bv;
            Cout[(size_t)(rbase + r) * H_ + col] = f2bf(val);
        }
    }
}

// ---------------------------------------------------------------------------
// Kernel 6: GEMM2  out = g_ft (2048 x 1024, bf16) @ cls_w + cls_b  (fp32 out)
// WG: 8 waves (2 along M, 4 along N); wave tile 16 x 64; N padded to 256.
// ---------------------------------------------------------------------------
__global__ __launch_bounds__(256) void gemm2(const uint16_t* __restrict__ A,   // gft
                                             const uint16_t* __restrict__ BT,  // clsT
                                             const float* __restrict__ bias,   // cls_b
                                             float* __restrict__ Out) {
    const int lane = threadIdx.x & 31;
    const int wave = threadIdx.x >> 5;
    const int wm = wave & 1;                 // 0..1
    const int wn = wave >> 1;                // 0..3
    const int m0 = blockIdx.x * 32 + wm * 16;
    const int n0 = wn * 64;
    const int l15 = lane & 15;
    const bool hi = lane >= 16;

    v8f acc[4] = {};
    const uint16_t* arow = A + (size_t)(m0 + l15) * H_;

    for (int k0 = 0; k0 < H_; k0 += 32) {
        Frag a;
        int ka = k0 + (hi ? 8 : 0);
        a.u[0] = *(const uint4*)(arow + ka);
        a.u[1] = *(const uint4*)(arow + ka + 16);
#pragma unroll
        for (int t = 0; t < 4; ++t) {
            const uint16_t* bcol = BT + (size_t)(n0 + t * 16 + l15) * H_ + k0 + (hi ? 8 : 0);
            Frag bf;
            bf.u[0] = *(const uint4*)(bcol);
            bf.u[1] = *(const uint4*)(bcol + 16);
            acc[t] = __builtin_amdgcn_wmma_f32_16x16x32_bf16(
                false, a.v, false, bf.v, (short)0, acc[t], false, false);
        }
    }

    const int rbase = m0 + (hi ? 8 : 0);
#pragma unroll
    for (int t = 0; t < 4; ++t) {
        int col = n0 + t * 16 + l15;
        if (col < C_) {
            float bv = bias[col];
#pragma unroll
            for (int r = 0; r < 8; ++r) {
                Out[(size_t)(rbase + r) * C_ + col] = acc[t][r] + bv;
            }
        }
    }
}

// ---------------------------------------------------------------------------
// Launch
// ---------------------------------------------------------------------------
extern "C" void kernel_launch(void* const* d_in, const int* in_sizes, int n_in,
                              void* d_out, int out_size, void* d_ws, size_t ws_size,
                              hipStream_t stream) {
    const float* part_feats = (const float*)d_in[0];
    const float* cdds       = (const float*)d_in[1];
    const float* fc_w       = (const float*)d_in[2];
    const float* fc_b       = (const float*)d_in[3];
    const float* cls_w      = (const float*)d_in[4];
    const float* cls_b      = (const float*)d_in[5];
    float* out = (float*)d_out;

    // Workspace layout (bytes)
    char* ws = (char*)d_ws;
    uint16_t* Q    = (uint16_t*)(ws);                        // 2048*4096*2 = 16,777,216
    uint16_t* fcwT = (uint16_t*)(ws + 16777216);             // 1024*4096*2 =  8,388,608
    uint16_t* gft  = (uint16_t*)(ws + 25165824);             // 2048*1024*2 =  4,194,304
    uint16_t* clsT = (uint16_t*)(ws + 29360128);             //  256*1024*2 =    524,288
    float*    coef = (float*)   (ws + 29884416);             // 2048*6*4    =     49,152

    // 1) per-batch weight row-sums
    coef_kernel<<<(B_ + 255) / 256, 256, 0, stream>>>(cdds, coef);

    // 2) stream part_feats once -> Q (bf16)
    build_q<<<(B_ * D_) / 256, 256, 0, stream>>>(part_feats, coef, Q);

    // 3) fc_w -> fcwT (bf16, transposed)
    transpose_fcw<<<dim3(K1 / 32, H_ / 32), dim3(32, 8), 0, stream>>>(fc_w, fcwT);

    // 4) cls_w -> clsT (bf16, transposed, zero-padded to 256 cols)
    build_clsT<<<(CP * H_) / 256, 256, 0, stream>>>(cls_w, clsT);

    // 5) g_ft = Q @ fc_w + fc_b        (WMMA bf16, M=2048 N=1024 K=4096)
    gemm1<<<dim3(B_ / 64, H_ / 128), 256, 0, stream>>>(Q, fcwT, fc_b, gft);

    // 6) out = g_ft @ cls_w + cls_b    (WMMA bf16, M=2048 N=256(200) K=1024)
    gemm2<<<dim3(B_ / 32, 1), 256, 0, stream>>>(gft, clsT, cls_b, out);
}
